// GraphTransformer_2448131359385
// MI455X (gfx1250) — compile-verified, matrix operand order checked by reading, not run on hardware
//
#include <hip/hip_runtime.h>
#include <hip/hip_bf16.h>
#include <math.h>

typedef __bf16 v16bf __attribute__((ext_vector_type(16)));
typedef __bf16 v8bf  __attribute__((ext_vector_type(8)));
typedef float  v8f   __attribute__((ext_vector_type(8)));

#define Ntok 4096      // E: sequence length / GCN node count
#define E2   65536
#define DE   200
#define DD   256
#define DFF  2048

// fragment-order LDS mapping:
//   lane = (row&15) + 16*((k>>3)&1) ; elem = ((k>>4)<<3) + (k&7)
// -> for a group of 8 consecutive k (kg = k>>3): laneAdd = (kg&1)<<4,
//    elemBase = (kg>>1)<<3, elems contiguous.

// ---------------------------------------------------------------------------
// Generic bf16-WMMA GEMM: C[M,N] = act(A[M,K] * B[K,N] + bias)
// block = 128 threads (4 waves), block tile 64x64, k-step 32.
// LDS tiles stored fragment-major: consumer does 2x ds_load_b128 per fragment.
// act: 0=none, 1=relu, 2=leaky(0.01)
// ---------------------------------------------------------------------------
__global__ __launch_bounds__(128)
void gemm_wmma_kernel(const float* __restrict__ A, const float* __restrict__ B,
                      const float* __restrict__ bias, float* __restrict__ C,
                      int M, int N, int K, int act) {
  __shared__ __align__(32) __bf16 As2[4 * 32 * 16];  // [rowTile][lane][elem]
  __shared__ __align__(32) __bf16 Bs2[4 * 32 * 16];  // [ntTile][lane][elem]
  const int tid  = threadIdx.x;
  const int lane = tid & 31;
  const int wave = tid >> 5;
  const int laneLo = lane & 15;
  const int hi     = lane >> 4;
  const int m0 = blockIdx.y * 64;
  const int n0 = blockIdx.x * 64;

  v8f acc0 = {}; v8f acc1 = {}; v8f acc2 = {}; v8f acc3 = {};
  const int Kp = (K + 31) & ~31;

  for (int k0 = 0; k0 < Kp; k0 += 32) {
    // ---- stage A: 64 rows x 32 k = 256 groups of 8; 2 groups per thread ----
#pragma unroll
    for (int gi = 0; gi < 2; ++gi) {
      int g = tid + gi * 128;
      int r = g >> 2, kg = g & 3;
      int gk = k0 + kg * 8;
      v8bf pk;
      if (gk < K) {  // K is a multiple of 8 -> whole group valid
        const float4* p = (const float4*)(A + (size_t)(m0 + r) * K + gk);
        float4 a = p[0], b = p[1];
        pk[0] = (__bf16)a.x; pk[1] = (__bf16)a.y; pk[2] = (__bf16)a.z; pk[3] = (__bf16)a.w;
        pk[4] = (__bf16)b.x; pk[5] = (__bf16)b.y; pk[6] = (__bf16)b.z; pk[7] = (__bf16)b.w;
      } else {
#pragma unroll
        for (int j = 0; j < 8; ++j) pk[j] = (__bf16)0.0f;
      }
      int dst = ((r >> 4) << 9) + (((r & 15) + ((kg & 1) << 4)) << 4) + ((kg >> 1) << 3);
      *(v8bf*)&As2[dst] = pk;
    }
    // ---- stage B: 32 k x 64 cols; gather 8 along k per thread ----
#pragma unroll
    for (int gi = 0; gi < 2; ++gi) {
      int g = tid + gi * 128;
      int c = g >> 2, kg = g & 3;
      int gk = k0 + kg * 8;
      v8bf pk;
      if (gk < K) {
        const float* p = B + (size_t)gk * N + n0 + c;
#pragma unroll
        for (int j = 0; j < 8; ++j) pk[j] = (__bf16)p[(size_t)j * N];
      } else {
#pragma unroll
        for (int j = 0; j < 8; ++j) pk[j] = (__bf16)0.0f;
      }
      int dst = ((c >> 4) << 9) + (((c & 15) + ((kg & 1) << 4)) << 4) + ((kg >> 1) << 3);
      *(v8bf*)&Bs2[dst] = pk;
    }
    __syncthreads();

    v16bf af = *(const v16bf*)&As2[(wave << 9) + (lane << 4)];
    v16bf b0 = *(const v16bf*)&Bs2[(0 << 9) + (lane << 4)];
    v16bf b1 = *(const v16bf*)&Bs2[(1 << 9) + (lane << 4)];
    v16bf b2 = *(const v16bf*)&Bs2[(2 << 9) + (lane << 4)];
    v16bf b3 = *(const v16bf*)&Bs2[(3 << 9) + (lane << 4)];
    acc0 = __builtin_amdgcn_wmma_f32_16x16x32_bf16(false, af, false, b0, (short)0, acc0, false, false);
    acc1 = __builtin_amdgcn_wmma_f32_16x16x32_bf16(false, af, false, b1, (short)0, acc1, false, false);
    acc2 = __builtin_amdgcn_wmma_f32_16x16x32_bf16(false, af, false, b2, (short)0, acc2, false, false);
    acc3 = __builtin_amdgcn_wmma_f32_16x16x32_bf16(false, af, false, b3, (short)0, acc3, false, false);
    __syncthreads();
  }

#pragma unroll
  for (int nt = 0; nt < 4; ++nt) {
    v8f acc = (nt == 0) ? acc0 : (nt == 1) ? acc1 : (nt == 2) ? acc2 : acc3;
#pragma unroll
    for (int r = 0; r < 8; ++r) {
      int row = m0 + wave * 16 + hi * 8 + r;
      int col = n0 + nt * 16 + laneLo;
      float v = acc[r];
      if (bias) v += bias[col];
      if (act == 1) v = fmaxf(v, 0.0f);
      else if (act == 2) v = (v > 0.0f) ? v : 0.01f * v;
      C[(size_t)row * N + col] = v;
    }
  }
}

// ---------------------------------------------------------------------------
// Flash attention with WMMA. qkv: [S, 768] (q|k|v, head-major 8x32 inside).
// Block = 4 waves = 64 queries of one head; K/V tiles (32 keys) are staged
// cooperatively into fragment-ordered LDS once per block per tile.
// S^T = K*Q^T -> lane=query, vgpr=key -> softmax via shfl_xor(16), and the
// exp'd P lands exactly in A-fragment layout for P*V.
// ---------------------------------------------------------------------------
__global__ __launch_bounds__(128)
void flash_attn_kernel(const float* __restrict__ qkv, float* __restrict__ out, int S) {
  __shared__ __align__(32) __bf16 Ks2[2 * 32 * 16];  // key subtiles (A layout)
  __shared__ __align__(32) __bf16 Vs2[2 * 32 * 16];  // dim subtiles (B layout)
  const int tid    = threadIdx.x;
  const int lane   = tid & 31;
  const int wave   = tid >> 5;
  const int laneLo = lane & 15;
  const int hi     = lane >> 4;
  const int head   = blockIdx.y;
  const int q0     = (blockIdx.x * 4 + wave) * 16;
  const float scale = 0.17677669529663687f;  // 1/sqrt(32)

  v16bf qa;  // Q fragment, loaded once (also valid as B operand of K*Q^T)
  {
    const float* qp = qkv + (size_t)(q0 + laneLo) * 768 + head * 32;
#pragma unroll
    for (int i = 0; i < 16; ++i) {
      int t = i >> 1;
      int kk = ((t & 4) ? 16 : 0) + hi * 8 + ((t & 3) << 1) + (i & 1);
      qa[i] = (__bf16)(qp[kk] * scale);
    }
  }

  float m_run = -1e30f, l_run = 0.0f;
  v8f o0 = {}; v8f o1 = {};

  const int stg_r  = tid >> 2;   // 0..31
  const int stg_kg = tid & 3;    // 0..3

  for (int k0 = 0; k0 < S; k0 += 32) {
    // ---- stage K tile (32 keys x 32 dh), coalesced float4 reads ----
    {
      const float4* p = (const float4*)(qkv + (size_t)(k0 + stg_r) * 768 + 256 + head * 32 + stg_kg * 8);
      float4 a = p[0], b = p[1];
      v8bf pk;
      pk[0] = (__bf16)a.x; pk[1] = (__bf16)a.y; pk[2] = (__bf16)a.z; pk[3] = (__bf16)a.w;
      pk[4] = (__bf16)b.x; pk[5] = (__bf16)b.y; pk[6] = (__bf16)b.z; pk[7] = (__bf16)b.w;
      int dst = ((stg_r >> 4) << 9) + (((stg_r & 15) + ((stg_kg & 1) << 4)) << 4) + ((stg_kg >> 1) << 3);
      *(v8bf*)&Ks2[dst] = pk;
    }
    // ---- stage V tile (32 keys x 32 dims), gather along keys (B layout) ----
    {
      const float* p = qkv + (size_t)(k0 + stg_kg * 8) * 768 + 512 + head * 32 + stg_r;
      v8bf pk;
#pragma unroll
      for (int j = 0; j < 8; ++j) pk[j] = (__bf16)p[(size_t)j * 768];
      int dst = ((stg_r >> 4) << 9) + (((stg_r & 15) + ((stg_kg & 1) << 4)) << 4) + ((stg_kg >> 1) << 3);
      *(v8bf*)&Vs2[dst] = pk;
    }
    __syncthreads();

    v16bf ka0 = *(const v16bf*)&Ks2[(lane << 4)];
    v16bf ka1 = *(const v16bf*)&Ks2[512 + (lane << 4)];
    v16bf vb0 = *(const v16bf*)&Vs2[(lane << 4)];
    v16bf vb1 = *(const v16bf*)&Vs2[512 + (lane << 4)];

    v8f s0 = {}; v8f s1 = {};
    s0 = __builtin_amdgcn_wmma_f32_16x16x32_bf16(false, ka0, false, qa, (short)0, s0, false, false);
    s1 = __builtin_amdgcn_wmma_f32_16x16x32_bf16(false, ka1, false, qa, (short)0, s1, false, false);

    // this lane holds query=laneLo, keys k0+hi*8+r (s0) and k0+16+hi*8+r (s1)
    float tmax = -1e30f;
#pragma unroll
    for (int r = 0; r < 8; ++r) tmax = fmaxf(tmax, fmaxf(s0[r], s1[r]));
    tmax = fmaxf(tmax, __shfl_xor(tmax, 16, 32));
    float m_new = fmaxf(m_run, tmax);
    float corr  = __expf(m_run - m_new);

    float p0[8], p1[8], tsum = 0.0f;
#pragma unroll
    for (int r = 0; r < 8; ++r) {
      p0[r] = __expf(s0[r] - m_new);
      p1[r] = __expf(s1[r] - m_new);
      tsum += p0[r] + p1[r];
    }
    tsum += __shfl_xor(tsum, 16, 32);
    l_run = l_run * corr + tsum;
    m_run = m_new;

    v16bf pa;  // already A-fragment layout for this lane
#pragma unroll
    for (int i = 0; i < 8; ++i) { pa[i] = (__bf16)p0[i]; pa[i + 8] = (__bf16)p1[i]; }

    float cq[8];
#pragma unroll
    for (int r = 0; r < 8; ++r) cq[r] = __shfl(corr, r + hi * 8, 32);
#pragma unroll
    for (int r = 0; r < 8; ++r) { o0[r] *= cq[r]; o1[r] *= cq[r]; }

    o0 = __builtin_amdgcn_wmma_f32_16x16x32_bf16(false, pa, false, vb0, (short)0, o0, false, false);
    o1 = __builtin_amdgcn_wmma_f32_16x16x32_bf16(false, pa, false, vb1, (short)0, o1, false, false);
    __syncthreads();
  }

  float li[8];
#pragma unroll
  for (int r = 0; r < 8; ++r) li[r] = 1.0f / __shfl(l_run, r + hi * 8, 32);
#pragma unroll
  for (int r = 0; r < 8; ++r) {
    int row = q0 + r + hi * 8;
    out[(size_t)row * 256 + head * 32 + laneLo]      = o0[r] * li[r];
    out[(size_t)row * 256 + head * 32 + 16 + laneLo] = o1[r] * li[r];
  }
}

// ---------------------------------------------------------------------------
// Front-end / GCN / norm / head kernels
// ---------------------------------------------------------------------------
__global__ void dist_kernel(const int* __restrict__ ei, const float* __restrict__ pos,
                            float* __restrict__ dist) {
  int e = blockIdx.x * 256 + threadIdx.x;
  if (e >= Ntok) return;
  int r = ei[e] - 1, c = ei[Ntok + e] - 1;
  float dx = pos[c * 3 + 0] - pos[r * 3 + 0];
  float dy = pos[c * 3 + 1] - pos[r * 3 + 1];
  float dz = pos[c * 3 + 2] - pos[r * 3 + 2];
  dist[e] = sqrtf(dx * dx + dy * dy + dz * dz);
}

__global__ __launch_bounds__(256)
void dist_stats_kernel(const float* __restrict__ dist, float* __restrict__ st) {
  __shared__ float rs[256], rq[256];
  int tid = threadIdx.x;
  float s = 0.0f, q = 0.0f;
  for (int e = tid; e < Ntok; e += 256) { float v = dist[e]; s += v; q += v * v; }
  rs[tid] = s; rq[tid] = q; __syncthreads();
  for (int st2 = 128; st2 > 0; st2 >>= 1) {
    if (tid < st2) { rs[tid] += rs[tid + st2]; rq[tid] += rq[tid + st2]; }
    __syncthreads();
  }
  if (tid == 0) { float m = rs[0] / Ntok; st[0] = m; st[1] = rq[0] / Ntok - m * m; }
}

// tok[e][j] = emb[x[row]][j] + emb[x[col]][j] + BN(dist[e]*w[j]) (rank-1 BN closed form)
__global__ void token_kernel(const int* __restrict__ ei, const int* __restrict__ xid,
                             const float* __restrict__ emb, const float* __restrict__ dist,
                             const float* __restrict__ st, const float* __restrict__ w,
                             const float* __restrict__ g, const float* __restrict__ b,
                             float* __restrict__ tok) {
  int e = blockIdx.x;
  int r = xid[ei[e] - 1];
  int c = xid[ei[Ntok + e] - 1];
  float dd = dist[e] - st[0];
  float var_d = st[1];
  for (int j = threadIdx.x; j < DE; j += 256) {
    float wj = w[j];
    float ef = dd * wj * rsqrtf(var_d * wj * wj + 1e-6f) * g[j] + b[j];
    tok[(size_t)e * DE + j] = emb[(size_t)r * DE + j] + emb[(size_t)c * DE + j] + ef;
  }
}

__global__ void dinv_init_kernel(float* deg) {
  int i = blockIdx.x * 256 + threadIdx.x;
  if (i < Ntok) deg[i] = 1.0f;  // self loop
}
__global__ void deg_scatter_kernel(const int* __restrict__ y, float* deg) {
  int e = blockIdx.x * 256 + threadIdx.x;
  if (e < E2) atomicAdd(&deg[y[E2 + e]], 1.0f);
}
__global__ void dinv_fin_kernel(float* deg) {
  int i = blockIdx.x * 256 + threadIdx.x;
  if (i < Ntok) deg[i] = rsqrtf(deg[i]);
}

__global__ void gcn_init_kernel(const float* __restrict__ xw, const float* __restrict__ dinv,
                                const float* __restrict__ bias, float* __restrict__ out) {
  size_t idx = (size_t)blockIdx.x * 256 + threadIdx.x;
  int i = (int)(idx >> 8), d = (int)(idx & 255);
  out[idx] = bias[d] + xw[idx] * dinv[i] * dinv[i];
}

__global__ __launch_bounds__(64)
void gcn_scatter_kernel(const float* __restrict__ xw, const int* __restrict__ y,
                        const float* __restrict__ dinv, float* __restrict__ out) {
  int e = blockIdx.x;
  int s = y[e], t = y[E2 + e];
  float w = dinv[s] * dinv[t];
  for (int d = threadIdx.x; d < DD; d += 64)
    atomicAdd(&out[(size_t)t * DD + d], xw[(size_t)s * DD + d] * w);
}

// out = leaky(a + (b? b : 0))
__global__ void add_leaky_kernel(const float* __restrict__ a, const float* __restrict__ b,
                                 float* __restrict__ out, int n) {
  int i = blockIdx.x * 256 + threadIdx.x;
  if (i >= n) return;
  float v = a[i] + (b ? b[i] : 0.0f);
  out[i] = (v > 0.0f) ? v : 0.01f * v;
}

__global__ __launch_bounds__(256)
void bn_stats_kernel(const float* __restrict__ x, int rows, int cols, float* __restrict__ mv) {
  __shared__ float rs[256], rq[256];
  int col = blockIdx.x, tid = threadIdx.x;
  float s = 0.0f, q = 0.0f;
  for (int r = tid; r < rows; r += 256) {
    float v = x[(size_t)r * cols + col];
    s += v; q += v * v;
  }
  rs[tid] = s; rq[tid] = q; __syncthreads();
  for (int st = 128; st > 0; st >>= 1) {
    if (tid < st) { rs[tid] += rs[tid + st]; rq[tid] += rq[tid + st]; }
    __syncthreads();
  }
  if (tid == 0) { float m = rs[0] / rows; mv[col] = m; mv[cols + col] = rq[0] / rows - m * m; }
}

__global__ void bn_apply_kernel(const float* __restrict__ x, const float* __restrict__ mv,
                                const float* __restrict__ g, const float* __restrict__ b,
                                float* __restrict__ out, int rows, int cols) {
  size_t idx = (size_t)blockIdx.x * 256 + threadIdx.x;
  if (idx >= (size_t)rows * cols) return;
  int col = (int)(idx % cols);
  out[idx] = (x[idx] - mv[col]) * rsqrtf(mv[cols + col] + 1e-6f) * g[col] + b[col];
}

__global__ __launch_bounds__(256)
void ln_res_kernel(const float* __restrict__ x, const float* __restrict__ res,
                   const float* __restrict__ g, const float* __restrict__ b,
                   float* __restrict__ out) {
  __shared__ float red[256];
  int row = blockIdx.x, tid = threadIdx.x;
  float v = x[(size_t)row * 256 + tid] + res[(size_t)row * 256 + tid];
  red[tid] = v; __syncthreads();
  for (int s = 128; s > 0; s >>= 1) { if (tid < s) red[tid] += red[tid + s]; __syncthreads(); }
  float mean = red[0] / 256.0f; __syncthreads();
  float d = v - mean;
  red[tid] = d * d; __syncthreads();
  for (int s = 128; s > 0; s >>= 1) { if (tid < s) red[tid] += red[tid + s]; __syncthreads(); }
  float var = red[0] / 256.0f;
  out[(size_t)row * 256 + tid] = d * rsqrtf(var + 1e-5f) * g[tid] + b[tid];
}

__global__ void heads_kernel(const float* __restrict__ h, const float* __restrict__ w3,
                             const float* __restrict__ b3, const float* __restrict__ w4,
                             const float* __restrict__ b4, float* __restrict__ out) {
  int idx = blockIdx.x * 256 + threadIdx.x;
  if (idx >= Ntok * 10) return;
  int e = idx / 10, c = idx % 10;
  const float* hr = h + (size_t)e * DD;
  if (c == 0) {
    float s = b3[0];
    for (int k = 0; k < DD; ++k) s += hr[k] * w3[k];
    out[e] = s;
  } else {
    int j = c - 1;
    float s = b4[j];
    for (int k = 0; k < DD; ++k) s += hr[k] * w4[k * 9 + j];
    out[Ntok + (size_t)e * 9 + j] = s;
  }
}

// ---------------------------------------------------------------------------
extern "C" void kernel_launch(void* const* d_in, const int* in_sizes, int n_in,
                              void* d_out, int out_size, void* d_ws, size_t ws_size,
                              hipStream_t stream) {
  (void)in_sizes; (void)n_in; (void)out_size; (void)ws_size;
  const int*   xid    = (const int*)d_in[0];
  const int*   ei     = (const int*)d_in[1];
  const float* pos    = (const float*)d_in[2];
  const int*   y      = (const int*)d_in[3];
  const float* emb    = (const float*)d_in[4];
  const float* fc1_w  = (const float*)d_in[5];
  const float* bn4_g  = (const float*)d_in[7];
  const float* bn4_b  = (const float*)d_in[8];
  const float* c1w    = (const float*)d_in[9];
  const float* c1b    = (const float*)d_in[10];
  const float* c2w    = (const float*)d_in[11];
  const float* c2b    = (const float*)d_in[12];
  const float* bn1_g  = (const float*)d_in[13];
  const float* bn1_b  = (const float*)d_in[14];
  const float* bn2_g  = (const float*)d_in[15];
  const float* bn2_b  = (const float*)d_in[16];
  const float* bn3_g  = (const float*)d_in[17];
  const float* bn3_b  = (const float*)d_in[18];
  const float* inw    = (const float*)d_in[19];
  const float* inb    = (const float*)d_in[20];
  const float* ow     = (const float*)d_in[21];
  const float* ob     = (const float*)d_in[22];
  const float* ln1g   = (const float*)d_in[23];
  const float* ln1b   = (const float*)d_in[24];
  const float* ln2g   = (const float*)d_in[25];
  const float* ln2b   = (const float*)d_in[26];
  const float* f1w    = (const float*)d_in[27];
  const float* f1b    = (const float*)d_in[28];
  const float* f2w    = (const float*)d_in[29];
  const float* f2b    = (const float*)d_in[30];
  const float* w3     = (const float*)d_in[31];
  const float* b3     = (const float*)d_in[32];
  const float* w4     = (const float*)d_in[33];
  const float* b4     = (const float*)d_in[34];

  float* ws    = (float*)d_ws;
  float* tok   = ws;                          // 4096*200
  float* h     = tok + (size_t)Ntok * DE;     // 4096*256
  float* t1    = h   + (size_t)Ntok * DD;
  float* t2    = t1  + (size_t)Ntok * DD;
  float* big   = t2  + (size_t)Ntok * DD;     // 4096*2048
  float* dist  = big + (size_t)Ntok * DFF;    // 4096
  float* dinv  = dist + Ntok;                 // 4096
  float* dstat = dinv + Ntok;                 // 2
  float* colmv = dstat + 2;                   // 2*256

  const int nED = Ntok * DD;   // 1048576

  // ---- edge features -> tokens ----
  dist_kernel<<<Ntok / 256, 256, 0, stream>>>(ei, pos, dist);
  dist_stats_kernel<<<1, 256, 0, stream>>>(dist, dstat);
  token_kernel<<<Ntok, 256, 0, stream>>>(ei, xid, emb, dist, dstat, fc1_w, bn4_g, bn4_b, tok);

  // ---- degrees for GCN graph y ----
  dinv_init_kernel<<<Ntok / 256, 256, 0, stream>>>(dinv);
  deg_scatter_kernel<<<E2 / 256, 256, 0, stream>>>(y, dinv);
  dinv_fin_kernel<<<Ntok / 256, 256, 0, stream>>>(dinv);

  // ---- GCN 1: leaky(gcn(tok)), bn1 ----
  gemm_wmma_kernel<<<dim3(DD / 64, Ntok / 64), 128, 0, stream>>>(tok, c1w, nullptr, t1, Ntok, DD, DE, 0);
  gcn_init_kernel<<<nED / 256, 256, 0, stream>>>(t1, dinv, c1b, t2);
  gcn_scatter_kernel<<<E2, 64, 0, stream>>>(t1, y, dinv, t2);
  add_leaky_kernel<<<nED / 256, 256, 0, stream>>>(t2, nullptr, t2, nED);
  bn_stats_kernel<<<DD, 256, 0, stream>>>(t2, Ntok, DD, colmv);
  bn_apply_kernel<<<nED / 256, 256, 0, stream>>>(t2, colmv, bn1_g, bn1_b, h, Ntok, DD);

  // ---- GCN 2: leaky(h + gcn(h)), bn2 ----
  gemm_wmma_kernel<<<dim3(DD / 64, Ntok / 64), 128, 0, stream>>>(h, c2w, nullptr, t1, Ntok, DD, DD, 0);
  gcn_init_kernel<<<nED / 256, 256, 0, stream>>>(t1, dinv, c2b, t2);
  gcn_scatter_kernel<<<E2, 64, 0, stream>>>(t1, y, dinv, t2);
  add_leaky_kernel<<<nED / 256, 256, 0, stream>>>(h, t2, h, nED);
  bn_stats_kernel<<<DD, 256, 0, stream>>>(h, Ntok, DD, colmv);
  bn_apply_kernel<<<nED / 256, 256, 0, stream>>>(h, colmv, bn2_g, bn2_b, h, Ntok, DD);

  // ---- transformer layers ----
  for (int i = 0; i < 2; ++i) {
    gemm_wmma_kernel<<<dim3(768 / 64, Ntok / 64), 128, 0, stream>>>(
        h, inw + (size_t)i * DD * 768, inb + (size_t)i * 768, big, Ntok, 768, DD, 0);
    flash_attn_kernel<<<dim3(Ntok / 64, 8), 128, 0, stream>>>(big, t1, Ntok);
    gemm_wmma_kernel<<<dim3(DD / 64, Ntok / 64), 128, 0, stream>>>(
        t1, ow + (size_t)i * DD * DD, ob + (size_t)i * DD, t2, Ntok, DD, DD, 0);
    ln_res_kernel<<<Ntok, 256, 0, stream>>>(h, t2, ln1g + i * DD, ln1b + i * DD, h);
    gemm_wmma_kernel<<<dim3(DFF / 64, Ntok / 64), 128, 0, stream>>>(
        h, f1w + (size_t)i * DD * DFF, f1b + (size_t)i * DFF, big, Ntok, DFF, DD, 1);
    gemm_wmma_kernel<<<dim3(DD / 64, Ntok / 64), 128, 0, stream>>>(
        big, f2w + (size_t)i * DFF * DD, f2b + (size_t)i * DD, t1, Ntok, DD, DFF, 0);
    ln_res_kernel<<<Ntok, 256, 0, stream>>>(h, t1, ln2g + i * DD, ln2b + i * DD, h);
  }

  // ---- tail: leaky, bn3, heads ----
  add_leaky_kernel<<<nED / 256, 256, 0, stream>>>(h, nullptr, h, nED);
  bn_stats_kernel<<<DD, 256, 0, stream>>>(h, Ntok, DD, colmv);
  bn_apply_kernel<<<nED / 256, 256, 0, stream>>>(h, colmv, bn3_g, bn3_b, h, Ntok, DD);
  heads_kernel<<<(Ntok * 10 + 255) / 256, 256, 0, stream>>>(h, w3, b3, w4, b4, (float*)d_out);
}
